// SwinTransformerBlock3D_49331994361925
// MI455X (gfx1250) — matrix operand order repaired and checked by reading, not compiled, not run on hardware
//
#include <hip/hip_runtime.h>
#include <hip/hip_bf16.h>
#include <math.h>

typedef _Float16 v16h __attribute__((ext_vector_type(16)));
typedef float    v8f  __attribute__((ext_vector_type(8)));
typedef unsigned int u32x4 __attribute__((ext_vector_type(4)));
typedef int          i32x4 __attribute__((ext_vector_type(4)));
typedef int          i32x8 __attribute__((ext_vector_type(8)));

#if defined(__has_builtin)
#  if __has_builtin(__builtin_amdgcn_tensor_load_to_lds)
#    define HAVE_TDM 1
#  endif
#endif
#ifndef HAVE_TDM
#  define HAVE_TDM 0
#endif

// ---------------------------------------------------------------------------
// WMMA helpers (CDNA5 gfx1250, wave32): D(16x16,f32) = A(16x32,f16)*B(32x16,f16)+C
// ---------------------------------------------------------------------------
__device__ __forceinline__ v8f wmma_f16(v16h a, v16h b, v8f c) {
  return __builtin_amdgcn_wmma_f32_16x16x32_f16(false, a, false, b, (short)0, c, false, false);
}

// A fragment from row-major (M x K) f16, tile (m_base, k_base).
// Lane L: M = m_base+(L&15); halves 0..7 -> K = k_base+(L>>4)*8+0..7, halves 8..15 -> +16.
__device__ __forceinline__ v16h load_frag_a(const _Float16* __restrict__ A, int ldk,
                                            int m_base, int k_base, int lane) {
  const _Float16* p = A + (size_t)(m_base + (lane & 15)) * ldk + k_base + ((lane >> 4) << 3);
  v16h r;
  ((float4*)&r)[0] = *(const float4*)p;
  ((float4*)&r)[1] = *(const float4*)(p + 16);
  return r;
}

// B fragment from N-major (N x K) f16 (B^T storage), tile (k_base, n_base).
// Lane L: N = n_base+(L&15); halves 0..7 -> K = k_base+(L>>4)*8+0..7, halves 8..15 -> +16.
__device__ __forceinline__ v16h load_frag_b(const _Float16* __restrict__ BT, int ldk,
                                            int n_base, int k_base, int lane) {
  const _Float16* p = BT + (size_t)(n_base + (lane & 15)) * ldk + k_base + ((lane >> 4) << 3);
  v16h r;
  ((float4*)&r)[0] = *(const float4*)p;
  ((float4*)&r)[1] = *(const float4*)(p + 16);
  return r;
}

// ---------------------------------------------------------------------------
// Tensor Data Mover: 2D tile (td0 x td1 halves, row stride td0, contiguous in
// global) -> LDS, optionally padding 4 DWORDs (8 halves) after each
// (1<<pad_interval)*8 bytes to de-conflict LDS banks for fragment reads.
// ---------------------------------------------------------------------------
#if HAVE_TDM
__device__ __forceinline__ void tdm_load_2d(void* lds_dst, const void* gsrc,
                                            unsigned td0, unsigned td1, int pad_interval) {
  unsigned lds_off = (unsigned)(size_t)lds_dst;            // low 32 bits = LDS offset
  unsigned long long ga = (unsigned long long)(size_t)gsrc;
  unsigned w0 = 0x10000u;                                  // data_size = 1 (2-byte)
  if (pad_interval >= 0)
    w0 |= (1u << 20) | ((unsigned)pad_interval << 22) | (3u << 25);  // pad_amount=4 DW
  u32x4 g0 = { 1u, lds_off, (unsigned)ga, (unsigned)(ga >> 32) | (2u << 30) };
  i32x8 g1;
  g1[0] = (int)w0;
  g1[1] = (int)((td0 & 0xFFFFu) << 16);                       // tensor_dim0 lo16
  g1[2] = (int)((td0 >> 16) | ((td1 & 0xFFFFu) << 16));       // td0 hi | tensor_dim1 lo
  g1[3] = (int)((td1 >> 16) | (td0 << 16));                   // td1 hi | tile_dim0
  g1[4] = (int)td1;                                           // tile_dim1 (tile_dim2=0)
  g1[5] = (int)td0;                                           // tensor_dim0_stride lo
  g1[6] = 0; g1[7] = 0;
  i32x4 z4 = {0, 0, 0, 0};
#if __clang_major__ >= 23
  i32x8 z8 = {0, 0, 0, 0, 0, 0, 0, 0};
  __builtin_amdgcn_tensor_load_to_lds(g0, g1, z4, z4, z8, 0);
#else
  __builtin_amdgcn_tensor_load_to_lds(g0, g1, z4, z4, 0);
#endif
}
#else
__device__ __forceinline__ void coop_copy_pad(_Float16* dst, const _Float16* src,
                                              int td0, int td1, int ldp, int tid, int nthr) {
  int total = (td0 * td1) >> 3;
  for (int idx = tid; idx < total; idx += nthr) {
    int e = idx << 3;
    int r = e / td0, c = e % td0;
    *(float4*)(dst + r * ldp + c) = *(const float4*)(src + (size_t)r * td0 + c);
  }
}
#endif

// ---------------------------------------------------------------------------
// Geometry: T=8,H=64,W=64,C=128; window (4,8,8) -> N=256 tokens, 128 windows.
// roll(-2,-4,-4) folded into source index; inverse roll is the same map on output.
// ---------------------------------------------------------------------------

// Kernel 1: convert + transpose weights to f16, N-major.
__global__ void k_wconv(const float* __restrict__ qkv_w, const float* __restrict__ proj_w,
                        const float* __restrict__ fc1_w, const float* __restrict__ fc2_w,
                        _Float16* __restrict__ qkvT, _Float16* __restrict__ projT,
                        _Float16* __restrict__ fc1T, _Float16* __restrict__ fc2T) {
  int i = blockIdx.x * blockDim.x + threadIdx.x;           // 0..65535
  if (i < 384 * 128) { int n = i >> 7, k = i & 127; qkvT[i] = (_Float16)qkv_w[k * 384 + n]; }
  if (i < 128 * 128) { int n = i >> 7, k = i & 127; projT[i] = (_Float16)proj_w[k * 128 + n]; }
  if (i < 512 * 128) { int n = i >> 7, k = i & 127; fc1T[i]  = (_Float16)fc1_w[k * 512 + n]; }
  if (i < 128 * 512) { int n = i >> 9, k = i & 511; fc2T[i]  = (_Float16)fc2_w[k * 128 + n]; }
}

// Kernel 2: LN1 + roll + window partition -> xw f16 (windowed token-major).
__global__ void k_ln1(const float* __restrict__ x, const float* __restrict__ g,
                      const float* __restrict__ b, _Float16* __restrict__ xw) {
  int tok = blockIdx.x;                     // win*256 + n
  int win = tok >> 8, n = tok & 255;
  int wt = win >> 6, wh = (win >> 3) & 7, ww = win & 7;
  int lt = n >> 6,  lh = (n >> 3) & 7,   lw = n & 7;
  int st = ((wt * 4 + lt) + 2) & 7;
  int sh = ((wh * 8 + lh) + 4) & 63;
  int sw = ((ww * 8 + lw) + 4) & 63;
  const float* xp = x + (((size_t)st * 64 + sh) * 64 + sw) * 128;
  int c = threadIdx.x;
  float v = xp[c];
  float s = v, s2 = v * v;
  for (int o = 16; o > 0; o >>= 1) { s += __shfl_down(s, o); s2 += __shfl_down(s2, o); }
  __shared__ float rs[4], rs2[4];
  int wv = threadIdx.x >> 5;
  if ((threadIdx.x & 31) == 0) { rs[wv] = s; rs2[wv] = s2; }
  __syncthreads();
  float mu  = (rs[0] + rs[1] + rs[2] + rs[3]) * (1.0f / 128.0f);
  float var = (rs2[0] + rs2[1] + rs2[2] + rs2[3]) * (1.0f / 128.0f) - mu * mu;
  float r = rsqrtf(var + 1e-5f);
  xw[(size_t)tok * 128 + c] = (_Float16)((v - mu) * r * g[c] + b[c]);
}

// Kernel 3: QKV GEMM (32768x128 @ 128x384), branch-free epilogues.
// Q scaled by hd^-0.5, Q/K padded to hd=32 (zeros); V stored transposed.
__global__ void k_qkv(const _Float16* __restrict__ xw, const _Float16* __restrict__ qkvT,
                      const float* __restrict__ qkv_b,
                      _Float16* __restrict__ qp, _Float16* __restrict__ kp,
                      _Float16* __restrict__ vT) {
  int wave = threadIdx.x >> 5, lane = threadIdx.x & 31;
  __builtin_prefetch(qkvT + ((size_t)threadIdx.x << 5), 0, 0);  // global_prefetch_b8
  int m_base = (blockIdx.x * 8 + wave) * 16;
  int win = m_base >> 8;
  int nq0 = (m_base & 255) + ((lane >> 4) << 3);   // token row base for acc[i]
  int nidx = lane & 15;
  v16h afr[4];
#pragma unroll
  for (int kt = 0; kt < 4; kt++) afr[kt] = load_frag_a(xw, 128, m_base, kt * 32, lane);

  // ---- Q ----
  for (int head = 0; head < 8; head++) {
    v16h bf[4];
#pragma unroll
    for (int kt = 0; kt < 4; kt++) bf[kt] = load_frag_b(qkvT, 128, head * 16, kt * 32, lane);
    v8f acc = {};
#pragma unroll
    for (int kt = 0; kt < 4; kt++) acc = wmma_f16(afr[kt], bf[kt], acc);
    float bb = qkv_b[head * 16 + nidx];
    size_t base = ((size_t)(win * 8 + head) * 256 + nq0) * 32;
#pragma unroll
    for (int i = 0; i < 8; i++) {
      qp[base + i * 32 + nidx]      = (_Float16)((acc[i] + bb) * 0.25f);
      qp[base + i * 32 + 16 + nidx] = (_Float16)0.0f;
    }
  }
  // ---- K ----
  for (int head = 0; head < 8; head++) {
    v16h bf[4];
#pragma unroll
    for (int kt = 0; kt < 4; kt++) bf[kt] = load_frag_b(qkvT, 128, 128 + head * 16, kt * 32, lane);
    v8f acc = {};
#pragma unroll
    for (int kt = 0; kt < 4; kt++) acc = wmma_f16(afr[kt], bf[kt], acc);
    float bb = qkv_b[128 + head * 16 + nidx];
    size_t base = ((size_t)(win * 8 + head) * 256 + nq0) * 32;
#pragma unroll
    for (int i = 0; i < 8; i++) {
      kp[base + i * 32 + nidx]      = (_Float16)(acc[i] + bb);
      kp[base + i * 32 + 16 + nidx] = (_Float16)0.0f;
    }
  }
  // ---- V (transposed: head-major hd x 256) ----
  for (int head = 0; head < 8; head++) {
    v16h bf[4];
#pragma unroll
    for (int kt = 0; kt < 4; kt++) bf[kt] = load_frag_b(qkvT, 128, 256 + head * 16, kt * 32, lane);
    v8f acc = {};
#pragma unroll
    for (int kt = 0; kt < 4; kt++) acc = wmma_f16(afr[kt], bf[kt], acc);
    float bb = qkv_b[256 + head * 16 + nidx];
    size_t base = ((size_t)(win * 8 + head) * 16 + nidx) * 256 + nq0;
#pragma unroll
    for (int i = 0; i < 8; i++) vT[base + i] = (_Float16)(acc[i] + bb);
  }
}

__device__ __forceinline__ int region_cnt(int gt, int gh, int gw) {
  int rt = gt < 4 ? 0 : (gt < 6 ? 1 : 2);
  int rh = gh < 56 ? 0 : (gh < 60 ? 1 : 2);
  int rw = gw < 56 ? 0 : (gw < 60 ? 1 : 2);
  return rt * 9 + rh * 3 + rw;
}

// LDS layout (dynamic) for k_attn, all offsets 16B aligned:
#define ATT_S_OFF   0            // 64 x 257 f32            = 65,792 B
#define ATT_P_OFF   65792        // 64 x 264 f16 (padded)   = 33,792 B
#define ATT_K_OFF   99584        // 256 x 40 f16 (padded)   = 20,480 B
#define ATT_V_OFF   120064       // 16 x 264 f16 (padded)   =  8,448 B
#define ATT_SMEM    128512

// Kernel 4: attention per (window, head, 64-row chunk).
// TDM stages K and V^T into LDS (bank-conflict-free padded rows); S = Q K^T via
// WMMA + rel-pos bias + shift mask -> softmax -> P(f16) -> P V via WMMA.
__global__ void k_attn(const _Float16* __restrict__ qp, const _Float16* __restrict__ kp,
                       const _Float16* __restrict__ vT, const float* __restrict__ rpb,
                       _Float16* __restrict__ attn) {
  extern __shared__ char smem[];
  float*    S  = (float*)(smem + ATT_S_OFF);
  _Float16* P  = (_Float16*)(smem + ATT_P_OFF);
  _Float16* Kl = (_Float16*)(smem + ATT_K_OFF);
  _Float16* Vl = (_Float16*)(smem + ATT_V_OFF);
  int bid = blockIdx.x;
  int chunk = bid & 3;                  // 64-row chunk of 256 queries
  int wh = bid >> 2;                    // win*8 + head
  int win = wh >> 3, head = wh & 7;
  int wave = threadIdx.x >> 5, lane = threadIdx.x & 31;
  const _Float16* qb = qp + (size_t)wh * 256 * 32;
  const _Float16* kb = kp + (size_t)wh * 256 * 32;
  const _Float16* vb = vT + (size_t)wh * 16 * 256;
  int wt = win >> 6, whh = (win >> 3) & 7, www = win & 7;

  // --- Stage K (256 rows x 32 halves, pad every 64B) and V^T (16 x 256, pad
  //     every 512B) into LDS via the Tensor Data Mover.
#if HAVE_TDM
  if (wave == 0) {
    tdm_load_2d(Kl, kb, 32, 256, 3);    // rows of 64B -> LDS stride 40 halves
    tdm_load_2d(Vl, vb, 256, 16, 6);    // rows of 512B -> LDS stride 264 halves
    __builtin_amdgcn_s_wait_tensorcnt(0);
  }
#else
  coop_copy_pad(Kl, kb, 32, 256, 40, threadIdx.x, 256);
  coop_copy_pad(Vl, vb, 256, 16, 264, threadIdx.x, 256);
#endif
  __syncthreads();

  // --- Phase 1: S tiles. wave -> mt = wave>>1; each wave does 8 of 16 N-tiles.
  int mt = wave >> 1;
  int ntb = (wave & 1) * 8;
  v16h a = load_frag_a(qb, 32, chunk * 64 + mt * 16, 0, lane);
#pragma unroll
  for (int nn = 0; nn < 8; nn++) {
    int nt = ntb + nn;
    v8f acc = {};
    acc = wmma_f16(a, load_frag_b(Kl, 40, nt * 16, 0, lane), acc);
    int n_k = nt * 16 + (lane & 15);
    int tk = n_k >> 6, hk = (n_k >> 3) & 7, wk = n_k & 7;
    int rk = region_cnt(wt * 4 + tk, whh * 8 + hk, www * 8 + wk);
#pragma unroll
    for (int i = 0; i < 8; i++) {
      int lr = mt * 16 + i + ((lane >> 4) << 3);
      int n_q = chunk * 64 + lr;
      int tq = n_q >> 6, hq = (n_q >> 3) & 7, wq = n_q & 7;
      int rel = (tq - tk + 3) * 225 + (hq - hk + 7) * 15 + (wq - wk + 7);
      float v = acc[i] + rpb[rel * 8 + head];
      int rq = region_cnt(wt * 4 + tq, whh * 8 + hq, www * 8 + wq);
      if (rq != rk) v -= 100.0f;
      S[lr * 257 + n_k] = v;
    }
  }
  __syncthreads();

  // --- Phase 2: softmax, 4 threads per row (64 elems each), shfl reduce.
  {
    int row = threadIdx.x >> 2, part = threadIdx.x & 3;
    float* sr = S + row * 257 + part * 64;
    float m = -1e30f;
    for (int j = 0; j < 64; j++) m = fmaxf(m, sr[j]);
    m = fmaxf(m, __shfl_xor(m, 1));
    m = fmaxf(m, __shfl_xor(m, 2));
    float s = 0.0f;
    for (int j = 0; j < 64; j++) { float e = __expf(sr[j] - m); sr[j] = e; s += e; }
    s += __shfl_xor(s, 1);
    s += __shfl_xor(s, 2);
    float inv = 1.0f / s;
    _Float16* pr = P + row * 264 + part * 64;
    for (int j = 0; j < 64; j++) pr[j] = (_Float16)(sr[j] * inv);
  }
  __syncthreads();

  // --- Phase 3: out = P @ V (64x256 @ 256x16), waves 0..3, 8 chained WMMAs.
  if (wave < 4) {
    v8f acc = {};
#pragma unroll
    for (int kt = 0; kt < 8; kt++) {
      v16h pa  = load_frag_a((const _Float16*)P, 264, wave * 16, kt * 32, lane);
      v16h vb2 = load_frag_b(Vl, 264, 0, kt * 32, lane);
      acc = wmma_f16(pa, vb2, acc);
    }
    int col = head * 16 + (lane & 15);
#pragma unroll
    for (int i = 0; i < 8; i++) {
      int nq = chunk * 64 + wave * 16 + i + ((lane >> 4) << 3);
      attn[((size_t)win * 256 + nq) * 128 + col] = (_Float16)acc[i];
    }
  }
}

// Kernel 5: proj GEMM + window reverse + roll-back + residual -> x1 (f32).
__global__ void k_proj(const _Float16* __restrict__ attn, const _Float16* __restrict__ projT,
                       const float* __restrict__ proj_b, const float* __restrict__ x,
                       float* __restrict__ x1) {
  int wave = threadIdx.x >> 5, lane = threadIdx.x & 31;
  __builtin_prefetch(projT + ((size_t)threadIdx.x << 5), 0, 0);
  int m_base = (blockIdx.x * 8 + wave) * 16;
  v16h afr[4];
#pragma unroll
  for (int kt = 0; kt < 4; kt++) afr[kt] = load_frag_a(attn, 128, m_base, kt * 32, lane);
  for (int nt = 0; nt < 8; nt++) {
    v16h bf[4];
#pragma unroll
    for (int kt = 0; kt < 4; kt++) bf[kt] = load_frag_b(projT, 128, nt * 16, kt * 32, lane);
    v8f acc = {};
#pragma unroll
    for (int kt = 0; kt < 4; kt++) acc = wmma_f16(afr[kt], bf[kt], acc);
    int col = nt * 16 + (lane & 15);
    float pb = proj_b[col];
#pragma unroll
    for (int i = 0; i < 8; i++) {
      int m = m_base + i + ((lane >> 4) << 3);
      int win = m >> 8, n = m & 255;
      int wt = win >> 6, wh = (win >> 3) & 7, ww = win & 7;
      int lt = n >> 6,  lh = (n >> 3) & 7,   lw = n & 7;
      int ft = ((wt * 4 + lt) + 2) & 7;
      int fh = ((wh * 8 + lh) + 4) & 63;
      int fw = ((ww * 8 + lw) + 4) & 63;
      size_t idx = (((size_t)ft * 64 + fh) * 64 + fw) * 128 + col;
      x1[idx] = x[idx] + acc[i] + pb;
    }
  }
}

// Kernel 6: LN2 (natural layout) -> h2 f16.
__global__ void k_ln2(const float* __restrict__ x1, const float* __restrict__ g,
                      const float* __restrict__ b, _Float16* __restrict__ h2) {
  size_t tok = blockIdx.x;
  int c = threadIdx.x;
  float v = x1[tok * 128 + c];
  float s = v, s2 = v * v;
  for (int o = 16; o > 0; o >>= 1) { s += __shfl_down(s, o); s2 += __shfl_down(s2, o); }
  __shared__ float rs[4], rs2[4];
  int wv = threadIdx.x >> 5;
  if ((threadIdx.x & 31) == 0) { rs[wv] = s; rs2[wv] = s2; }
  __syncthreads();
  float mu  = (rs[0] + rs[1] + rs[2] + rs[3]) * (1.0f / 128.0f);
  float var = (rs2[0] + rs2[1] + rs2[2] + rs2[3]) * (1.0f / 128.0f) - mu * mu;
  float r = rsqrtf(var + 1e-5f);
  h2[tok * 128 + c] = (_Float16)((v - mu) * r * g[c] + b[c]);
}

// Kernel 7: fused MLP: FC1 + exact GELU into padded LDS, FC2 + residual -> out.
__global__ void k_mlp(const _Float16* __restrict__ h2, const _Float16* __restrict__ fc1T,
                      const float* __restrict__ fc1_b, const _Float16* __restrict__ fc2T,
                      const float* __restrict__ fc2_b, const float* __restrict__ x1,
                      float* __restrict__ out) {
  __shared__ _Float16 hid[32 * 520];               // 32 tokens x 512 hidden, padded
  int wave = threadIdx.x >> 5, lane = threadIdx.x & 31;
  __builtin_prefetch(fc1T + ((size_t)threadIdx.x << 5), 0, 0);
  __builtin_prefetch(fc2T + ((size_t)threadIdx.x << 5), 0, 0);
  int m_base = blockIdx.x * 32;
  // Phase 1: hidden = gelu(h2 @ fc1 + b1). 64 tiles, 8 per wave.
  for (int j = 0; j < 8; j++) {
    int t = wave * 8 + j;
    int mt = t >> 5, nt = t & 31;
    v16h bf[4];
#pragma unroll
    for (int kt = 0; kt < 4; kt++) bf[kt] = load_frag_b(fc1T, 128, nt * 16, kt * 32, lane);
    v8f acc = {};
#pragma unroll
    for (int kt = 0; kt < 4; kt++)
      acc = wmma_f16(load_frag_a(h2, 128, m_base + mt * 16, kt * 32, lane), bf[kt], acc);
    int col = nt * 16 + (lane & 15);
    float bb = fc1_b[col];
#pragma unroll
    for (int i = 0; i < 8; i++) {
      float v = acc[i] + bb;
      v = 0.5f * v * (1.0f + erff(v * 0.70710678f));
      int lr = mt * 16 + i + ((lane >> 4) << 3);
      hid[lr * 520 + col] = (_Float16)v;
    }
  }
  __syncthreads();
  // Phase 2: out = x1 + hidden @ fc2 + b2. 16 tiles, 2 per wave, K=512.
  for (int j = 0; j < 2; j++) {
    int t = wave * 2 + j;
    int mt = t >> 3, nt = t & 7;
    v8f acc = {};
#pragma unroll
    for (int kt = 0; kt < 16; kt++) {
      v16h a = load_frag_a((const _Float16*)hid, 520, mt * 16, kt * 32, lane);
      v16h b = load_frag_b(fc2T, 512, nt * 16, kt * 32, lane);
      acc = wmma_f16(a, b, acc);
    }
    int col = nt * 16 + (lane & 15);
    float bb = fc2_b[col];
#pragma unroll
    for (int i = 0; i < 8; i++) {
      int tok = m_base + mt * 16 + i + ((lane >> 4) << 3);
      size_t idx = (size_t)tok * 128 + col;
      out[idx] = x1[idx] + acc[i] + bb;
    }
  }
}

// ---------------------------------------------------------------------------
extern "C" void kernel_launch(void* const* d_in, const int* in_sizes, int n_in,
                              void* d_out, int out_size, void* d_ws, size_t ws_size,
                              hipStream_t stream) {
  (void)in_sizes; (void)n_in; (void)out_size; (void)ws_size;
  const float* x      = (const float*)d_in[0];
  const float* n1g    = (const float*)d_in[1];
  const float* n1b    = (const float*)d_in[2];
  const float* qkv_w  = (const float*)d_in[3];
  const float* qkv_b  = (const float*)d_in[4];
  const float* proj_w = (const float*)d_in[5];
  const float* proj_b = (const float*)d_in[6];
  const float* rpb    = (const float*)d_in[7];
  const float* n2g    = (const float*)d_in[8];
  const float* n2b    = (const float*)d_in[9];
  const float* fc1_w  = (const float*)d_in[10];
  const float* fc1_b  = (const float*)d_in[11];
  const float* fc2_w  = (const float*)d_in[12];
  const float* fc2_b  = (const float*)d_in[13];

  char* ws = (char*)d_ws;
  _Float16* xw    = (_Float16*)(ws + 0);            //  8,388,608 B
  _Float16* qp    = (_Float16*)(ws + 8388608);      // 16,777,216 B (hd padded to 32)
  _Float16* kp    = (_Float16*)(ws + 25165824);     // 16,777,216 B
  _Float16* vT    = (_Float16*)(ws + 41943040);     //  8,388,608 B
  _Float16* attn  = (_Float16*)(ws + 50331648);     //  8,388,608 B
  float*    x1    = (float*)   (ws + 58720256);     // 16,777,216 B
  _Float16* h2    = (_Float16*)(ws + 75497472);     //  8,388,608 B
  _Float16* qkvT  = (_Float16*)(ws + 83886080);     //     98,304 B
  _Float16* projT = (_Float16*)(ws + 83984384);     //     32,768 B
  _Float16* fc1T  = (_Float16*)(ws + 84017152);     //    131,072 B
  _Float16* fc2T  = (_Float16*)(ws + 84148224);     //    131,072 B  (total 84,279,296)

  k_wconv<<<256, 256, 0, stream>>>(qkv_w, proj_w, fc1_w, fc2_w, qkvT, projT, fc1T, fc2T);
  k_ln1<<<32768, 128, 0, stream>>>(x, n1g, n1b, xw);
  k_qkv<<<256, 256, 0, stream>>>(xw, qkvT, qkv_b, qp, kp, vT);
  k_attn<<<4096, 256, ATT_SMEM, stream>>>(qp, kp, vT, rpb, attn);
  k_proj<<<256, 256, 0, stream>>>(attn, projT, proj_b, x, x1);
  k_ln2<<<32768, 128, 0, stream>>>(x1, n2g, n2b, h2);
  k_mlp<<<1024, 256, 0, stream>>>(h2, fc1T, fc1_b, fc2T, fc2_b, x1, (float*)d_out);
}
